// CausalAttention_49589692400207
// MI455X (gfx1250) — matrix operand order repaired and checked by reading, not compiled, not run on hardware
//
#include <hip/hip_runtime.h>
#include <stdint.h>

typedef __attribute__((ext_vector_type(16))) __bf16 v16bf;
typedef __attribute__((ext_vector_type(8)))  float  v8f;

#define B_   8
#define N_   4096
#define CK_  64
#define CV_  128

__device__ __forceinline__ unsigned short f32_to_bf16(float f) {
  unsigned int u = __float_as_uint(f);
  u += 0x7fffu + ((u >> 16) & 1u);   // round-to-nearest-even
  return (unsigned short)(u >> 16);
}
__device__ __forceinline__ unsigned int pack_bf16x2(float a, float b) {
  return (unsigned int)f32_to_bf16(a) | ((unsigned int)f32_to_bf16(b) << 16);
}

// ---- prep: Q*(1/sqrt(Ck)) -> bf16; K -> bf16; V -> bf16 transposed [B][CV][N] ----
__global__ void cvt_kernel(const float* __restrict__ q,
                           const float* __restrict__ k,
                           const float* __restrict__ v,
                           unsigned short* __restrict__ qbf,
                           unsigned short* __restrict__ kbf,
                           unsigned short* __restrict__ vtbf) {
  const size_t nqk = (size_t)B_ * N_ * CK_;
  const size_t nv  = (size_t)B_ * N_ * CV_;
  size_t i = (size_t)blockIdx.x * blockDim.x + threadIdx.x;
  size_t stride = (size_t)gridDim.x * blockDim.x;
  for (size_t o = i; o < nqk; o += stride) {
    qbf[o] = f32_to_bf16(q[o] * 0.125f);   // fold 1/sqrt(64) into Q
    kbf[o] = f32_to_bf16(k[o]);
  }
  for (size_t o = i; o < nv; o += stride) {
    size_t b = o / ((size_t)CV_ * N_);
    size_t r = o % ((size_t)CV_ * N_);
    size_t c = r / N_;
    size_t n = r % N_;
    vtbf[o] = f32_to_bf16(v[(b * (size_t)N_ + n) * CV_ + c]);
  }
}

union U16 { v16bf v; uint4 q[2]; };

__device__ __forceinline__ void async_copy16(unsigned lds_byte_addr,
                                             unsigned glb_byte_off,
                                             const void* sbase) {
  // GLOBAL_LOAD_ASYNC_TO_LDS_B128, GVS mode: mem = SADDR(64) + VADDR(32) ; LDS = VDST VGPR
  asm volatile("global_load_async_to_lds_b128 %0, %1, %2"
               :: "v"(lds_byte_addr), "v"(glb_byte_off), "s"(sbase)
               : "memory");
}

__global__ __launch_bounds__(256)
void attn_kernel(const unsigned short* __restrict__ qbf,
                 const unsigned short* __restrict__ kbf,
                 const unsigned short* __restrict__ vtbf,
                 float* __restrict__ out) {
  __shared__ __align__(16) unsigned short ldsK[2][32 * 64];    // 2 x 4 KB, [key][feat]
  __shared__ __align__(16) unsigned short ldsV[2][128 * 32];   // 2 x 8 KB, [vcol][key]
  __shared__ __align__(16) unsigned short ldsP[8][16 * 32];    // 8 KB per-wave P tiles

  const int tid  = threadIdx.x;
  const int w    = tid >> 5;
  const int lane = tid & 31;
  const int half = lane >> 4;
  const int lcol = lane & 15;
  const int b    = blockIdx.x >> 5;
  const int qb   = blockIdx.x & 31;
  const int m0   = qb * 128 + w * 16;     // this wave's query-row base
  const int m0max = qb * 128 + 112;       // uniform per workgroup

  const float NEG_INF = -__builtin_inff();

  const unsigned short* Qb = qbf  + (size_t)b * N_ * CK_;
  const unsigned short* Kb = kbf  + (size_t)b * N_ * CK_;
  const unsigned short* Vb = vtbf + (size_t)b * CV_ * N_;

  // ---- Q tile as B-operand of S^T = K * Q^T : per-lane contiguous 32B column segments ----
  U16 bq[2];
#pragma unroll
  for (int c = 0; c < 2; ++c) {
    const unsigned short* p = Qb + (size_t)(m0 + lcol) * CK_ + c * 32 + 16 * half;
    bq[c].q[0] = *(const uint4*)p;
    bq[c].q[1] = *(const uint4*)(p + 8);
  }

  v8f zero = {};
  v8f acc[8];
#pragma unroll
  for (int vt = 0; vt < 8; ++vt) acc[vt] = zero;
  float m_s = NEG_INF, l_s = 0.0f;        // per-lane: softmax state of query (m0 + lcol)

  // prologue: stage block k0=0 into buffer 0
  {
    async_copy16((unsigned)(uintptr_t)&ldsK[0][tid * 8], (unsigned)(tid * 16), Kb);
#pragma unroll
    for (int j = 0; j < 2; ++j) {
      int c = tid * 2 + j, col = c >> 2, seg = c & 3;
      async_copy16((unsigned)(uintptr_t)&ldsV[0][col * 32 + seg * 8],
                   (unsigned)(col * N_ * 2 + seg * 16), Vb);
    }
  }

  for (int k0 = 0; k0 <= m0max; k0 += 32) {
    const int cur = (k0 >> 5) & 1;
    asm volatile("s_wait_asynccnt 0x0" ::: "memory");
    __syncthreads();

    // ---- pipeline: issue async copies for the next block into the other buffer ----
    if (k0 + 32 <= m0max) {
      const unsigned short* kbase = Kb + (size_t)(k0 + 32) * CK_;
      const unsigned short* vbase = Vb + (k0 + 32);
      async_copy16((unsigned)(uintptr_t)&ldsK[cur ^ 1][tid * 8], (unsigned)(tid * 16), kbase);
#pragma unroll
      for (int j = 0; j < 2; ++j) {
        int c = tid * 2 + j, col = c >> 2, seg = c & 3;
        async_copy16((unsigned)(uintptr_t)&ldsV[cur ^ 1][col * 32 + seg * 8],
                     (unsigned)(col * N_ * 2 + seg * 16), vbase);
      }
    }

    if (k0 <= m0) {
      // ---- S^T = K_tile(16x64) x Q^T(64x16): C rows = keys, cols = queries ----
      v8f s[2];
#pragma unroll
      for (int t = 0; t < 2; ++t) {
        const int n0 = k0 + t * 16;
        if (n0 <= m0) {
          U16 ak[2];   // K rows in A layout from LDS
#pragma unroll
          for (int c = 0; c < 2; ++c) {
            const unsigned short* p = &ldsK[cur][(t * 16 + lcol) * CK_ + c * 32 + 16 * half];
            ak[c].q[0] = *(const uint4*)p;
            ak[c].q[1] = *(const uint4*)(p + 8);
          }
          v8f sv = __builtin_amdgcn_wmma_f32_16x16x32_bf16(false, ak[0].v, false, bq[0].v,
                                                           (short)0, zero, false, false);
          sv = __builtin_amdgcn_wmma_f32_16x16x32_bf16(false, ak[1].v, false, bq[1].v,
                                                       (short)0, sv, false, false);
          if (n0 == m0) {   // diagonal: key j = n0 + r + 8*half, query i = m0 + lcol ; need j < i
#pragma unroll
            for (int r = 0; r < 8; ++r)
              if (!(r + 8 * half < lcol)) sv[r] = NEG_INF;
          }
          s[t] = sv;
        } else {
#pragma unroll
          for (int r = 0; r < 8; ++r) s[t][r] = NEG_INF;
        }
      }

      // ---- online softmax: in-lane reduce over 16 keys + one xor-16 exchange ----
      float x = fmaxf(s[0][0], s[1][0]);
#pragma unroll
      for (int r = 1; r < 8; ++r) x = fmaxf(x, fmaxf(s[0][r], s[1][r]));
      x = fmaxf(x, __shfl_xor(x, 16));
      const float mnew = fmaxf(m_s, x);
      const float corr = (mnew > NEG_INF) ? __expf(m_s - mnew) : 1.0f;
      float p0[8], p1[8];
      float ps = 0.0f;
#pragma unroll
      for (int r = 0; r < 8; ++r) {
        p0[r] = (s[0][r] > NEG_INF) ? __expf(s[0][r] - mnew) : 0.0f;
        p1[r] = (s[1][r] > NEG_INF) ? __expf(s[1][r] - mnew) : 0.0f;
        ps += p0[r] + p1[r];
      }
      ps += __shfl_xor(ps, 16);
      l_s = l_s * corr + ps;
      m_s = mnew;

      // ---- P^T scatter to LDS row-major [query][key0..31]: packed 64-bit stores ----
      unsigned short* lp = ldsP[w];
#pragma unroll
      for (int t = 0; t < 2; ++t) {
        const float* pp = t ? p1 : p0;
        uint2 w0, w1;
        w0.x = pack_bf16x2(pp[0], pp[1]); w0.y = pack_bf16x2(pp[2], pp[3]);
        w1.x = pack_bf16x2(pp[4], pp[5]); w1.y = pack_bf16x2(pp[6], pp[7]);
        unsigned short* dp = lp + lcol * 32 + t * 16 + 8 * half;
        *(uint2*)dp       = w0;   // keys +0..3
        *(uint2*)(dp + 4) = w1;   // keys +4..7
      }
      // ---- gather P as A-operand (16 queries x 32 keys) ----
      U16 ap;
      {
        const unsigned short* rp = lp + lcol * 32 + 16 * half;
        ap.q[0] = *(const uint4*)rp;
        ap.q[1] = *(const uint4*)(rp + 8);
      }
      // redistribute per-query corr into accumulator C-layout (row = r + 8*half)
      float cr[8];
#pragma unroll
      for (int r = 0; r < 8; ++r) cr[r] = __shfl(corr, r + 8 * half);

      // ---- O += P(16x32) x V(32x128): 8 N-tiles from LDS ----
#pragma unroll
      for (int vt = 0; vt < 8; ++vt) {
        U16 bv;
        const unsigned short* p = &ldsV[cur][(vt * 16 + lcol) * 32 + 16 * half];
        bv.q[0] = *(const uint4*)p;
        bv.q[1] = *(const uint4*)(p + 8);
#pragma unroll
        for (int r = 0; r < 8; ++r) acc[vt][r] *= cr[r];
        acc[vt] = __builtin_amdgcn_wmma_f32_16x16x32_bf16(false, ap.v, false, bv.v,
                                                          (short)0, acc[vt], false, false);
      }
    }
  }

  // ---- normalize and store (query 0 of canvas: l==0 -> zeros, matches reference) ----
  const float rl = (l_s > 0.0f) ? 1.0f / l_s : 0.0f;
  float rcp[8];
#pragma unroll
  for (int r = 0; r < 8; ++r) rcp[r] = __shfl(rl, r + 8 * half);
  float* Ob = out + ((size_t)b * N_ + m0) * CV_;
#pragma unroll
  for (int vt = 0; vt < 8; ++vt) {
#pragma unroll
    for (int r = 0; r < 8; ++r) {
      Ob[(size_t)(r + 8 * half) * CV_ + vt * 16 + lcol] = acc[vt][r] * rcp[r];
    }
  }
}

extern "C" void kernel_launch(void* const* d_in, const int* in_sizes, int n_in,
                              void* d_out, int out_size, void* d_ws, size_t ws_size,
                              hipStream_t stream) {
  const float* q = (const float*)d_in[0];
  const float* k = (const float*)d_in[1];
  const float* v = (const float*)d_in[2];
  float* out = (float*)d_out;

  // workspace: Q_bf16 (4MB) | K_bf16 (4MB) | V^T_bf16 (8MB) = 16 MB
  char* ws = (char*)d_ws;
  unsigned short* qbf  = (unsigned short*)ws;
  unsigned short* kbf  = (unsigned short*)(ws + (size_t)B_ * N_ * CK_ * 2);
  unsigned short* vtbf = (unsigned short*)(ws + (size_t)B_ * N_ * CK_ * 4);

  cvt_kernel<<<2048, 256, 0, stream>>>(q, k, v, qbf, kbf, vtbf);
  attn_kernel<<<B_ * 32, 256, 0, stream>>>(qbf, kbf, vtbf, out);
}
